// MessageBlock_23596550324905
// MI455X (gfx1250) — matrix operand order repaired and checked by reading, not compiled, not run on hardware
//
#include <hip/hip_runtime.h>

// MI455X / gfx1250 MessageBlock. All 4 matmuls on v_wmma_f32_16x16x32_bf16.
// Weights/activations pre-converted to bf16 once per launch; every WMMA
// operand is loaded as aligned 16-byte chunks directly into fragment regs.
// Kernel A processes 32 edges per wave (two M=16 A-tiles sharing each B
// fragment) to halve B-operand fetch bandwidth per WMMA, and the B fragment
// is software-pipelined (load b[nt+1] before the WMMAs consuming b[nt]) so
// fetch latency overlaps matrix math inside a single wave.
//
// Fragment layouts (CDNA5 ISA section 7.12.2):
//   A (16x32, 16-bit): lane L (M = L%16), elem j -> K = j + 16*(L/16) (order used here)
//                      => two contiguous 8-half runs at k0 and k0+8
//   B (32x16, 16-bit): lane L (N = L%16), elem j -> K = j + 16*(L/16) (order used here, same as A)
//                      => one contiguous 16-half run
//   C/D (16x16, f32) : reg r, lane L -> M = r + 8*(L/16), N = L%16

typedef __attribute__((ext_vector_type(16))) __bf16 v16bf;
typedef __attribute__((ext_vector_type(8)))  float  v8f;

#define HD 128
#define NN 100000
#define NE 600000

union frag16 { uint4 u[2]; v16bf v; };

// 16 contiguous bf16 (32 B, 16B-aligned) -> fragment
static __device__ inline v16bf lb16(const __bf16* __restrict__ p) {
  frag16 t;
  t.u[0] = *(const uint4*)(p);
  t.u[1] = *(const uint4*)(p + 8);
  return t.v;
}
// two 8-half runs (A-fragment)
static __device__ inline v16bf lbA(const __bf16* __restrict__ p0,
                                   const __bf16* __restrict__ p1) {
  frag16 t;
  t.u[0] = *(const uint4*)(p0);
  t.u[1] = *(const uint4*)(p1);
  return t.v;
}

static __device__ inline float sigmoidf_(float v) { return 1.0f / (1.0f + __expf(-v)); }

// ---------------- pre-conversion kernels ----------------
__global__ void cvt_bf16_kernel(const float* __restrict__ src, __bf16* __restrict__ dst, int n) {
  int i = blockIdx.x * 256 + threadIdx.x;
  int stride = gridDim.x * 256;
  for (; i < n; i += stride) dst[i] = (__bf16)src[i];
}
// W1 is [128][257]; keep only k<256 (last column handled as rank-1 term)
__global__ void cvt_w1_kernel(const float* __restrict__ W1, __bf16* __restrict__ w1b) {
  int i = blockIdx.x * 256 + threadIdx.x;
  if (i < 128 * 256) {
    int n = i >> 8, k = i & 255;
    w1b[i] = (__bf16)W1[n * 257 + k];
  }
}

// ---------------- Kernel A: edge MLP + scatter-add ----------------
// one wave per 32-edge tile (two M=16 sub-tiles); 8 waves / block
__global__ __launch_bounds__(256) void edge_mlp_kernel(
    const __bf16* __restrict__ xb, const int* __restrict__ eidx,
    const float* __restrict__ ea, const float* __restrict__ W1,
    const float* __restrict__ b1, const __bf16* __restrict__ w1b,
    const __bf16* __restrict__ w2b, const float* __restrict__ b2,
    float* __restrict__ agg) {
  __shared__ alignas(16) __bf16 h1s[8][32][HD];  // per-wave layer-1 restage (64 KB)

  const int lane = threadIdx.x & 31;
  const int w    = threadIdx.x >> 5;
  const int tile = blockIdx.x * 8 + w;
  if (tile >= NE / 32) return;
  const int e0  = tile * 32;
  const int l16 = lane & 15;
  const int hi  = lane >> 4;

  const int* rowp = eidx;       // edge_index[0]
  const int* colp = eidx + NE;  // edge_index[1]

  // edges feeding this lane's A rows (sub-tile s: M = l16, edge e0 + s*16 + l16)
  const __bf16* xr0 = xb + (size_t)rowp[e0 + l16] * HD;
  const __bf16* xc0 = xb + (size_t)colp[e0 + l16] * HD;
  const __bf16* xr1 = xb + (size_t)rowp[e0 + 16 + l16] * HD;
  const __bf16* xc1 = xb + (size_t)colp[e0 + 16 + l16] * HD;

  // ---- layer 1: [32 x 256] @ [256 x 128] ----
  v8f acc[2][8];
#pragma unroll
  for (int s = 0; s < 2; ++s)
#pragma unroll
    for (int t = 0; t < 8; ++t)
#pragma unroll
      for (int j = 0; j < 8; ++j) acc[s][t][j] = 0.0f;

  for (int kt = 0; kt < 8; ++kt) {
    const int k0 = kt * 32 + hi * 16;
    const int k1 = k0 + 8;
    v16bf a0 = lbA((k0 < HD) ? (xr0 + k0) : (xc0 + (k0 - HD)),
                   (k1 < HD) ? (xr0 + k1) : (xc0 + (k1 - HD)));
    v16bf a1 = lbA((k0 < HD) ? (xr1 + k0) : (xc1 + (k0 - HD)),
                   (k1 < HD) ? (xr1 + k1) : (xc1 + (k1 - HD)));
    const __bf16* wp = w1b + (size_t)l16 * 256 + kt * 32 + hi * 16;
    v16bf b = lb16(wp);  // B[k][n] = W1[n][k]
#pragma unroll
    for (int nt = 0; nt < 8; ++nt) {
      v16bf bn;
      if (nt < 7) bn = lb16(wp + (size_t)(nt + 1) * (16 * 256));  // prefetch next B
      acc[0][nt] = __builtin_amdgcn_wmma_f32_16x16x32_bf16(false, a0, false, b, (short)0,
                                                           acc[0][nt], false, false);
      acc[1][nt] = __builtin_amdgcn_wmma_f32_16x16x32_bf16(false, a1, false, b, (short)0,
                                                           acc[1][nt], false, false);
      if (nt < 7) b = bn;
    }
  }

  // per-row edge_attr + destination indices (sub-tile s, M = r + 8*hi)
  float eav[2][8];
  int dstn[2][8];
#pragma unroll
  for (int s = 0; s < 2; ++s)
#pragma unroll
    for (int r = 0; r < 8; ++r) {
      int e = e0 + s * 16 + r + 8 * hi;
      eav[s][r]  = ea[e];
      dstn[s][r] = rowp[e];
    }

  // rank-1 edge_attr column + bias + SiLU, stage to LDS as bf16
#pragma unroll
  for (int nt = 0; nt < 8; ++nt) {
    int n = nt * 16 + l16;
    float wl = W1[(size_t)n * 257 + 256];
    float bb = b1[n];
#pragma unroll
    for (int s = 0; s < 2; ++s)
#pragma unroll
      for (int r = 0; r < 8; ++r) {
        float v = acc[s][nt][r] + eav[s][r] * wl + bb;
        v = v * sigmoidf_(v);  // SiLU
        h1s[w][s * 16 + r + 8 * hi][n] = (__bf16)v;
      }
  }
  // LDS ops are in-order within a wave; each wave touches only its own slice.

  // ---- layer 2: [32 x 128] @ [128 x 128] ----
  v8f acc2[2][8];
#pragma unroll
  for (int s = 0; s < 2; ++s)
#pragma unroll
    for (int t = 0; t < 8; ++t)
#pragma unroll
      for (int j = 0; j < 8; ++j) acc2[s][t][j] = 0.0f;

  const __bf16* hp0 = &h1s[w][l16][0];
  const __bf16* hp1 = &h1s[w][16 + l16][0];
  for (int kt = 0; kt < 4; ++kt) {
    const int k0 = kt * 32 + hi * 16;
    v16bf a0 = lbA(hp0 + k0, hp0 + k0 + 8);
    v16bf a1 = lbA(hp1 + k0, hp1 + k0 + 8);
    const __bf16* wp = w2b + (size_t)l16 * HD + kt * 32 + hi * 16;
    v16bf b = lb16(wp);  // B[k][n] = W2[n][k]
#pragma unroll
    for (int nt = 0; nt < 8; ++nt) {
      v16bf bn;
      if (nt < 7) bn = lb16(wp + (size_t)(nt + 1) * (16 * HD));
      acc2[0][nt] = __builtin_amdgcn_wmma_f32_16x16x32_bf16(false, a0, false, b, (short)0,
                                                            acc2[0][nt], false, false);
      acc2[1][nt] = __builtin_amdgcn_wmma_f32_16x16x32_bf16(false, a1, false, b, (short)0,
                                                            acc2[1][nt], false, false);
      if (nt < 7) b = bn;
    }
  }

  // bias + scatter-add to agg[row[e]]
#pragma unroll
  for (int nt = 0; nt < 8; ++nt) {
    int n = nt * 16 + l16;
    float bb = b2[n];
#pragma unroll
    for (int s = 0; s < 2; ++s)
#pragma unroll
      for (int r = 0; r < 8; ++r) {
        atomicAdd(&agg[(size_t)dstn[s][r] * HD + n], acc2[s][nt][r] + bb);
      }
  }
}

// ---------------- Kernel B: GRU cell ----------------
// acc += A[16 nodes x 128] @ Wg[128 x 128].T slice (Wg row-major bf16, stride 128)
static __device__ inline void mm_acc8(v8f (&acc)[8], const __bf16* __restrict__ Arow,
                                      const __bf16* __restrict__ Wg, int l16, int hi) {
  for (int kt = 0; kt < 4; ++kt) {
    const int k0 = kt * 32 + hi * 16;
    v16bf a = lbA(Arow + k0, Arow + k0 + 8);
    const __bf16* wp = Wg + (size_t)l16 * HD + kt * 32 + hi * 16;
    v16bf b = lb16(wp);
#pragma unroll
    for (int nt = 0; nt < 8; ++nt) {
      v16bf bn;
      if (nt < 7) bn = lb16(wp + (size_t)(nt + 1) * (16 * HD));
      acc[nt] = __builtin_amdgcn_wmma_f32_16x16x32_bf16(false, a, false, b, (short)0,
                                                        acc[nt], false, false);
      if (nt < 7) b = bn;
    }
  }
}

__global__ __launch_bounds__(256) void gru_kernel(
    const float* __restrict__ x, const __bf16* __restrict__ xb,
    const __bf16* __restrict__ aggb,
    const __bf16* __restrict__ wihb, const __bf16* __restrict__ whhb,
    const float* __restrict__ bih, const float* __restrict__ bhh,
    float* __restrict__ out) {
  const int lane = threadIdx.x & 31;
  const int w    = threadIdx.x >> 5;
  const int tile = blockIdx.x * 8 + w;
  if (tile >= NN / 16) return;
  const int n0  = tile * 16;
  const int l16 = lane & 15;
  const int hi  = lane >> 4;

  const __bf16* ax = xb   + (size_t)(n0 + l16) * HD;  // A rows from x
  const __bf16* ag = aggb + (size_t)(n0 + l16) * HD;  // A rows from agg

  v8f hn[8], inn[8], g[8];
#pragma unroll
  for (int t = 0; t < 8; ++t)
#pragma unroll
    for (int j = 0; j < 8; ++j) { hn[t][j] = 0.0f; inn[t][j] = 0.0f; }

  mm_acc8(hn,  ax, whhb + 2 * HD * HD, l16, hi);  // h_n (pre-bias)
  mm_acc8(inn, ag, wihb + 2 * HD * HD, l16, hi);  // i_n (pre-bias)

  // r gate: sigmoid(agg@Wih_r.T + x@Whh_r.T + b), fused accumulation
#pragma unroll
  for (int t = 0; t < 8; ++t)
#pragma unroll
    for (int j = 0; j < 8; ++j) g[t][j] = 0.0f;
  mm_acc8(g, ag, wihb, l16, hi);
  mm_acc8(g, ax, whhb, l16, hi);
#pragma unroll
  for (int nt = 0; nt < 8; ++nt) {
    int n = nt * 16 + l16;
    float br  = bih[n] + bhh[n];
    float bin = bih[2 * HD + n];
    float bhn = bhh[2 * HD + n];
#pragma unroll
    for (int r = 0; r < 8; ++r) {
      float rv = sigmoidf_(g[nt][r] + br);
      inn[nt][r] = tanhf((inn[nt][r] + bin) + rv * (hn[nt][r] + bhn));  // n-gate
    }
  }

  // z gate, then output
#pragma unroll
  for (int t = 0; t < 8; ++t)
#pragma unroll
    for (int j = 0; j < 8; ++j) g[t][j] = 0.0f;
  mm_acc8(g, ag, wihb + 1 * HD * HD, l16, hi);
  mm_acc8(g, ax, whhb + 1 * HD * HD, l16, hi);
#pragma unroll
  for (int nt = 0; nt < 8; ++nt) {
    int n = nt * 16 + l16;
    float bz = bih[HD + n] + bhh[HD + n];
#pragma unroll
    for (int r = 0; r < 8; ++r) {
      float z = sigmoidf_(g[nt][r] + bz);
      int node = n0 + r + 8 * hi;
      float h = x[(size_t)node * HD + n];  // hidden state in f32 for the blend
      out[(size_t)node * HD + n] = (1.0f - z) * inn[nt][r] + z * h;
    }
  }
}

extern "C" void kernel_launch(void* const* d_in, const int* in_sizes, int n_in,
                              void* d_out, int out_size, void* d_ws, size_t ws_size,
                              hipStream_t stream) {
  const float* x    = (const float*)d_in[0];
  const int*   eidx = (const int*)d_in[1];   // edge_index [2,E] (int per harness contract)
  const float* ea   = (const float*)d_in[2];
  const float* W1   = (const float*)d_in[3];
  const float* b1   = (const float*)d_in[4];
  const float* W2   = (const float*)d_in[5];
  const float* b2   = (const float*)d_in[6];
  const float* Wih  = (const float*)d_in[7];
  const float* Whh  = (const float*)d_in[8];
  const float* bih  = (const float*)d_in[9];
  const float* bhh  = (const float*)d_in[10];
  float* out = (float*)d_out;

  // workspace layout (all offsets 16B-aligned)
  char* ws = (char*)d_ws;
  float*  agg  = (float*)(ws);                          // 51,200,000 B
  __bf16* xbuf = (__bf16*)(ws + 51200000);              // 25,600,000 B
  __bf16* aggb = (__bf16*)(ws + 76800000);              // 25,600,000 B
  __bf16* w1b  = (__bf16*)(ws + 102400000);             //     65,536 B
  __bf16* w2b  = (__bf16*)(ws + 102400000 + 65536);     //     32,768 B
  __bf16* wihb = (__bf16*)(ws + 102400000 + 98304);     //     98,304 B
  __bf16* whhb = (__bf16*)(ws + 102400000 + 196608);    //     98,304 B

  hipMemsetAsync(agg, 0, (size_t)NN * HD * sizeof(float), stream);

  // one-time bf16 pre-passes (bandwidth-trivial vs 23.3 TB/s)
  cvt_bf16_kernel<<<2048, 256, 0, stream>>>(x, xbuf, NN * HD);
  cvt_w1_kernel<<<(128 * 256 + 255) / 256, 256, 0, stream>>>(W1, w1b);
  cvt_bf16_kernel<<<64, 256, 0, stream>>>(W2, w2b, HD * HD);
  cvt_bf16_kernel<<<192, 256, 0, stream>>>(Wih, wihb, 3 * HD * HD);
  cvt_bf16_kernel<<<192, 256, 0, stream>>>(Whh, whhb, 3 * HD * HD);

  const int tilesA  = NE / 32;  // 18750
  const int blocksA = (tilesA + 7) / 8;
  edge_mlp_kernel<<<blocksA, 256, 0, stream>>>(xbuf, eidx, ea, W1, b1, w1b, w2b, b2, agg);

  cvt_bf16_kernel<<<2048, 256, 0, stream>>>(agg, aggb, NN * HD);

  const int tilesB  = NN / 16;  // 6250
  const int blocksB = (tilesB + 7) / 8;
  gru_kernel<<<blocksB, 256, 0, stream>>>(x, xbuf, aggb, wihb, whhb, bih, bhh, out);
}
